// DeepSeekMLA_62302795595935
// MI455X (gfx1250) — matrix operand order repaired and checked by reading, not compiled, not run on hardware
//
#include <hip/hip_runtime.h>

// ---------------------------------------------------------------------------
// DeepSeek MLA decode for MI455X (gfx1250), absorbed-latent formulation.
// Attention runs in RANK=512 latent space with bf16 WMMA (16x16x32).
// K/V tiles staged into LDS by the Tensor Data Mover (tensor_load_to_lds),
// overlapped with the in-wave transpose of the P@V operand.
// ---------------------------------------------------------------------------

#define B_     8
#define T_     4
#define S0_    4096
#define H_     16
#define Dh_    128
#define R_     64
#define C_     64
#define RANK_  512
#define DM_    2048
#define STOT_  4100          // S0 + T
#define TS_    64            // S tile
#define SPAD_  4160          // 65 tiles of 64
#define NTILE_ 65
#define KQ_    576           // RANK + R   (score contraction length)
#define QP_    584           // q row pitch (bf16 elems)
#define SRP_   520           // cKV tile pitch: 512 + 4-dword TDM pad
#define KRP_   72            // kR tile pitch: 64 + 4-dword TDM pad
#define RSP_   72            // [r][s] tile pitch
#define PP_    72            // P row pitch
#define SCP_   17            // score scratch pitch (f32)

#define CDIV(a,b) (((a)+(b)-1)/(b))

typedef __attribute__((ext_vector_type(16))) __bf16 v16bf;
typedef __attribute__((ext_vector_type(8)))  float  v8f;
typedef __attribute__((ext_vector_type(4)))  unsigned int u32x4;
typedef __attribute__((ext_vector_type(8)))  unsigned int u32x8;

union FragBF { v16bf v; uint4 q[2]; };
union Pack8  { unsigned short u[8]; uint4 q; };

__device__ __forceinline__ v8f wmma_bf16(const FragBF& a, const FragBF& b, v8f c) {
  // (neg_a, A, neg_b, B, c_mod, C, reuse_a, reuse_b)
  return __builtin_amdgcn_wmma_f32_16x16x32_bf16(false, a.v, false, b.v,
                                                 (short)0, c, false, false);
}

__device__ __forceinline__ unsigned short f2bf(float f) {
  union { __bf16 h; unsigned short u; } cv;
  cv.h = (__bf16)f;
  return cv.u;
}

// ---------------------------------------------------------------------------
// TDM: DMA a 2D tile (tile_d0 x tile_d1 elements of 2 bytes) from global to
// LDS, appending (pad_amount+1) dwords of padding after every
// 8*2^pad_interval bytes (one row) so LDS rows land on a conflict-free pitch.
// D# layout per CDNA5 ISA ยง8.3/8.4. Tracked by TENSORcnt.
// ---------------------------------------------------------------------------
__device__ __forceinline__ void tdm_load_2d_bf16(unsigned lds_addr,
                                                 const void* gaddr,
                                                 unsigned tensor_d0,
                                                 unsigned tensor_d1,
                                                 unsigned tile_d0,
                                                 unsigned tile_d1,
                                                 unsigned stride0,
                                                 unsigned pad_interval,
                                                 unsigned pad_amount) {
  unsigned long long ga = (unsigned long long)(size_t)gaddr;
  u32x4 g0;
  g0[0] = 1u;                                            // count=1 (valid user D#)
  g0[1] = lds_addr;                                      // LDS byte address
  g0[2] = (unsigned)ga;                                  // global_addr[31:0]
  g0[3] = (unsigned)((ga >> 32) & 0x01FFFFFFu)           // global_addr[56:32]
        | 0x80000000u;                                   // type=2 ("image")
  u32x8 g1;
  g1[0] = (1u << 16)                                     // data_size: 2 bytes
        | (1u << 20)                                     // pad_enable
        | (pad_interval << 22)                           // pad interval code
        | (pad_amount << 25);                            // pad amount code
  g1[1] = (tensor_d0 & 0xFFFFu) << 16;                   // tensor_dim0[15:0]
  g1[2] = (tensor_d0 >> 16) | ((tensor_d1 & 0xFFFFu) << 16);
  g1[3] = (tensor_d1 >> 16) | (tile_d0 << 16);           // tile_dim0
  g1[4] = tile_d1;                                       // tile_dim1 (tile_dim2=0)
  g1[5] = stride0;                                       // tensor_dim0_stride lo
  g1[6] = 0u;
  g1[7] = 0u;
  asm volatile("tensor_load_to_lds %0, %1" :: "s"(g0), "s"(g1) : "memory");
}

// ---------------------------------------------------------------------------
// K1: y[32 x N] = x[32 x 2048] @ W[2048 x N]   (memory-bound on W)
// ---------------------------------------------------------------------------
__global__ void k_proj(const float* __restrict__ x, const float* __restrict__ W,
                       float* __restrict__ y, int N) {
  int idx = blockIdx.x * blockDim.x + threadIdx.x;
  if (idx >= 32 * N) return;
  int row = idx / N, col = idx % N;
  const float* xr = x + (size_t)row * DM_;
  float s = 0.f;
  for (int k = 0; k < DM_; ++k) s += xr[k] * W[(size_t)k * N + col];
  y[idx] = s;
}

// ---------------------------------------------------------------------------
// K2: build absorbed+rope'd q in bf16, scale 1/sqrt(D) folded in.
//  qB[b][h][t(16 pad)][QP_]: cols 0..511 = q_content @ W_uk_h, 512..575 = rope(q_rope)
// ---------------------------------------------------------------------------
__global__ void k_build_q(const float* __restrict__ qc,   // [bt][H*C]
                          const float* __restrict__ qr,   // [bt][H*R]
                          const float* __restrict__ wup,  // [RANK][H*Dh]
                          unsigned short* __restrict__ qB) {
  int idx = blockIdx.x * blockDim.x + threadIdx.x;
  const int total = B_ * H_ * 16 * QP_;
  if (idx >= total) return;
  int col = idx % QP_;
  int t   = (idx / QP_) % 16;
  int h   = (idx / (QP_ * 16)) % H_;
  int b   = idx / (QP_ * 16 * H_);
  const float scale = 0.08838834764831845f;  // 1/sqrt(128)
  float val = 0.f;
  if (t < T_) {
    int bt = b * T_ + t;
    if (col < RANK_) {
      const float* q = qc + ((size_t)bt * H_ + h) * C_;
      const float* w = wup + (size_t)col * (H_ * Dh_) + h * Dh_;
      float s = 0.f;
      for (int c = 0; c < C_; ++c) s += q[c] * w[c];
      val = s * scale;
    } else if (col < KQ_) {
      int r  = col - RANK_;
      int rm = r & 31;
      const float* q = qr + ((size_t)bt * H_ + h) * R_;
      float inv = __powf(10000.f, -(float)rm / 32.f);
      float fr  = (float)(S0_ + t) * inv;
      float cs = __cosf(fr), sn = __sinf(fr);
      float x1 = q[rm], x2 = q[rm + 32];
      val = (r < 32 ? x1 * cs - x2 * sn : x1 * sn + x2 * cs) * scale;
    }
  }
  qB[idx] = f2bf(val);
}

// ---------------------------------------------------------------------------
// K3a: cKVb[b][SPAD][RANK] bf16  (cache ++ new, zero padded)
// ---------------------------------------------------------------------------
__global__ void k_build_ckv(const float* __restrict__ cache,
                            const float* __restrict__ cnew,   // [bt][RANK]
                            unsigned short* __restrict__ dst) {
  int g = blockIdx.x * blockDim.x + threadIdx.x;
  const int total = B_ * SPAD_ * (RANK_ / 8);
  if (g >= total) return;
  int rc = g % (RANK_ / 8);
  int s  = (g / (RANK_ / 8)) % SPAD_;
  int b  = g / ((RANK_ / 8) * SPAD_);
  Pack8 p;
  if (s < S0_) {
    const float* src = cache + ((size_t)b * S0_ + s) * RANK_ + rc * 8;
    for (int j = 0; j < 8; ++j) p.u[j] = f2bf(src[j]);
  } else if (s < STOT_) {
    const float* src = cnew + ((size_t)(b * T_ + (s - S0_))) * RANK_ + rc * 8;
    for (int j = 0; j < 8; ++j) p.u[j] = f2bf(src[j]);
  } else {
    for (int j = 0; j < 8; ++j) p.u[j] = 0;
  }
  *(uint4*)(dst + (size_t)g * 8) = p.q;
}

// ---------------------------------------------------------------------------
// K3b: kRb[b][h][SPAD][R] bf16  (transpose cache + rope'd new rows, zero pad)
// ---------------------------------------------------------------------------
__global__ void k_build_kr(const float* __restrict__ ckr,    // [b][s][h][r]
                           const float* __restrict__ krraw,  // [bt][H*R]
                           unsigned short* __restrict__ dst) {
  int g = blockIdx.x * blockDim.x + threadIdx.x;
  const int total = B_ * H_ * SPAD_ * (R_ / 8);
  if (g >= total) return;
  int rc = g % (R_ / 8);
  int s  = (g / (R_ / 8)) % SPAD_;
  int h  = (g / ((R_ / 8) * SPAD_)) % H_;
  int b  = g / ((R_ / 8) * SPAD_ * H_);
  Pack8 p;
  if (s < S0_) {
    const float* src = ckr + (((size_t)b * S0_ + s) * H_ + h) * R_ + rc * 8;
    for (int j = 0; j < 8; ++j) p.u[j] = f2bf(src[j]);
  } else if (s < STOT_) {
    int t = s - S0_;
    const float* raw = krraw + ((size_t)(b * T_ + t) * H_ + h) * R_;
    for (int j = 0; j < 8; ++j) {
      int r  = rc * 8 + j;
      int rm = r & 31;
      float inv = __powf(10000.f, -(float)rm / 32.f);
      float fr  = (float)s * inv;           // position == s for new rows
      float cs = __cosf(fr), sn = __sinf(fr);
      float x1 = raw[rm], x2 = raw[rm + 32];
      p.u[j] = f2bf(r < 32 ? x1 * cs - x2 * sn : x1 * sn + x2 * cs);
    }
  } else {
    for (int j = 0; j < 8; ++j) p.u[j] = 0;
  }
  *(uint4*)(dst + (size_t)g * 8) = p.q;
}

// ---------------------------------------------------------------------------
// K4: flash attention in latent space. 1 WG per (b,h), 256 threads = 8 waves.
//   scores^T (s x t) = cKV_tile(s x 512) @ q_abs^T + kR_tile(s x 64) @ q_rope^T
//   O^T      (r x t) += cKV_tile^T(r x s) @ P^T       (all waves, 64 r each)
// cKV/kR tiles are DMA'd into LDS by the TDM while all waves build the
// transposed cKV^T operand; TENSORcnt gates consumption.
// ---------------------------------------------------------------------------
// LDS layout (bytes)
#define OFF_Q   0                          // 16 x QP_        (q, bf16)
#define OFF_SR  18688                      // 64 x SRP_       (cKV tile)
#define OFF_KR  (OFF_SR + 64 * SRP_ * 2)   // 64 x KRP_       (kR tile)
#define OFF_RS  (OFF_KR + 64 * KRP_ * 2)   // 512 x RSP_      (cKV^T tile)
#define OFF_P   (OFF_RS + RANK_ * RSP_ * 2)// 16 x PP_        (P, bf16)
#define OFF_SC  (OFF_P + 16 * PP_ * 2)     // 64 x SCP_ f32   (raw scores)
#define OFF_ST  (OFF_SC + TS_ * SCP_ * 4)  // m/l/alpha[16] f32
#define LDSB    (OFF_ST + 3 * 16 * 4)      // 175040 bytes

__global__ __launch_bounds__(256) void k_mla_attn(
    const unsigned short* __restrict__ qBg,
    const unsigned short* __restrict__ cKVg,
    const unsigned short* __restrict__ kRg,
    float* __restrict__ oLat) {
  extern __shared__ char smem[];
  __bf16* s_q  = (__bf16*)(smem + OFF_Q);
  __bf16* s_sr = (__bf16*)(smem + OFF_SR);
  __bf16* s_kr = (__bf16*)(smem + OFF_KR);
  __bf16* s_rs = (__bf16*)(smem + OFF_RS);
  __bf16* s_p  = (__bf16*)(smem + OFF_P);
  float*  s_sc = (float*)(smem + OFF_SC);
  float*  s_m  = (float*)(smem + OFF_ST);
  float*  s_l  = s_m + 16;
  float*  s_a  = s_m + 32;

  const int tid  = threadIdx.x;
  const int wave = tid >> 5;
  const int lane = tid & 31;
  const int nlo  = lane & 15;   // t column / A-row-in-tile
  const int hi   = lane >> 4;
  const int bh   = blockIdx.x;
  const int b    = bh / H_;

  const __bf16* qg   = (const __bf16*)qBg + (size_t)bh * 16 * QP_;
  const __bf16* ckvb = (const __bf16*)cKVg + (size_t)b * SPAD_ * RANK_;
  const __bf16* krb  = (const __bf16*)kRg + (size_t)bh * SPAD_ * R_;

  // LDS byte offsets for the TDM descriptors
  const unsigned lds_sr = (unsigned)(size_t)s_sr;
  const unsigned lds_kr = (unsigned)(size_t)s_kr;

  // load q tile (16 x QP_) once
  for (int i = tid; i < 16 * (QP_ * 2 / 16); i += 256)
    ((uint4*)s_q)[i] = ((const uint4*)qg)[i];
  if (tid < 16) { s_m[tid] = -1e30f; s_l[tid] = 0.f; }

  v8f acc[4] = {};

  for (int it = 0; it < NTILE_; ++it) {
    const int s0 = it * TS_;
    __syncthreads();   // previous tiles / P fully consumed

    // ---- TDM: DMA cKV tile [64][512] -> s_sr (pitch 520) and
    //           kR  tile [64][64]  -> s_kr (pitch 72), wave 0 only.
    if (wave == 0) {
      // pad_interval=7 (256 dwords = one 512-elem row), pad_amount=3 (4 dwords)
      tdm_load_2d_bf16(lds_sr, ckvb + (size_t)s0 * RANK_,
                       RANK_, SPAD_ - s0, RANK_, TS_, RANK_, 7u, 3u);
      // pad_interval=4 (32 dwords = one 64-elem row), pad_amount=3 (4 dwords)
      tdm_load_2d_bf16(lds_kr, krb + (size_t)s0 * R_,
                       R_, SPAD_ - s0, R_, TS_, R_, 4u, 3u);
    }

    // ---- tile_rs [r=512][s=64]: transposed cKV tile (overlaps the TDM)
    for (int g = tid; g < 1024; g += 256) {
      const int sq = g >> 6;            // s quad
      const int rc = g & 63;            // r chunk of 8
      const __bf16* p0 = ckvb + (size_t)(s0 + sq * 4) * RANK_ + rc * 8;
      uint4 r0 = *(const uint4*)(p0);
      uint4 r1 = *(const uint4*)(p0 + RANK_);
      uint4 r2 = *(const uint4*)(p0 + 2 * RANK_);
      uint4 r3 = *(const uint4*)(p0 + 3 * RANK_);
      const unsigned short* e0 = (const unsigned short*)&r0;
      const unsigned short* e1 = (const unsigned short*)&r1;
      const unsigned short* e2 = (const unsigned short*)&r2;
      const unsigned short* e3 = (const unsigned short*)&r3;
      for (int j = 0; j < 8; ++j) {
        unsigned long long v = (unsigned long long)e0[j]
                             | ((unsigned long long)e1[j] << 16)
                             | ((unsigned long long)e2[j] << 32)
                             | ((unsigned long long)e3[j] << 48);
        *(unsigned long long*)(s_rs + (rc * 8 + j) * RSP_ + sq * 4) = v;
      }
    }

    if (wave == 0) __builtin_amdgcn_s_wait_tensorcnt(0);
    __syncthreads();

    // ---- scores^T: waves 0-3 each own 16 s-rows, K = 512 (cKV) + 64 (rope)
    if (wave < 4) {
      v8f sc = {};
      const __bf16* arow = s_sr + (wave * 16 + nlo) * SRP_;
      const __bf16* krow = s_kr + (wave * 16 + nlo) * KRP_;
      const __bf16* brow = s_q + nlo * QP_;
      for (int ks = 0; ks < RANK_ / 32; ++ks) {
        FragBF a, bf;
        a.q[0]  = *(const uint4*)(arow + ks * 32 + hi * 8);
        a.q[1]  = *(const uint4*)(arow + ks * 32 + hi * 8 + 16);
        bf.q[0] = *(const uint4*)(brow + ks * 32 + hi * 8);
        bf.q[1] = *(const uint4*)(brow + ks * 32 + hi * 8 + 16);
        sc = wmma_bf16(a, bf, sc);
      }
      for (int ks = 0; ks < R_ / 32; ++ks) {
        FragBF a, bf;
        a.q[0]  = *(const uint4*)(krow + ks * 32 + hi * 8);
        a.q[1]  = *(const uint4*)(krow + ks * 32 + hi * 8 + 16);
        bf.q[0] = *(const uint4*)(brow + RANK_ + ks * 32 + hi * 8);
        bf.q[1] = *(const uint4*)(brow + RANK_ + ks * 32 + hi * 8 + 16);
        sc = wmma_bf16(a, bf, sc);
      }
      for (int v = 0; v < 8; ++v)
        s_sc[(wave * 16 + v + hi * 8) * SCP_ + nlo] = sc[v];
    }
    __syncthreads();

    // ---- online softmax update (16 threads, one per t)
    if (tid < 16) {
      const int t = tid;
      int valid = STOT_ - s0; if (valid > TS_) valid = TS_;
      float mOld = s_m[t], mNew = mOld;
      for (int s = 0; s < valid; ++s) mNew = fmaxf(mNew, s_sc[s * SCP_ + t]);
      float alpha = __expf(mOld - mNew);
      float sum = 0.f;
      for (int s = 0; s < TS_; ++s) {
        float p = 0.f;
        if (s < valid) p = __expf(s_sc[s * SCP_ + t] - mNew);
        s_p[t * PP_ + s] = (__bf16)p;
        sum += p;
      }
      s_l[t] = s_l[t] * alpha + sum;
      s_m[t] = mNew;
      s_a[t] = alpha;
    }
    __syncthreads();

    // ---- rescale accumulators + O^T += cKV^T @ P^T (each wave: 64 r cols)
    const float alpha = s_a[nlo];
    for (int i = 0; i < 4; ++i) acc[i] = acc[i] * alpha;
    for (int ks = 0; ks < 2; ++ks) {
      FragBF bf;
      const __bf16* prow = s_p + nlo * PP_ + ks * 32 + hi * 8;
      bf.q[0] = *(const uint4*)(prow);
      bf.q[1] = *(const uint4*)(prow + 16);
      for (int i = 0; i < 4; ++i) {
        const __bf16* arow = s_rs + ((wave * 4 + i) * 16 + nlo) * RSP_ + ks * 32;
        FragBF a;
        a.q[0] = *(const uint4*)(arow + hi * 8);
        a.q[1] = *(const uint4*)(arow + hi * 8 + 16);
        acc[i] = wmma_bf16(a, bf, acc[i]);
      }
    }
  }

  __syncthreads();
  const float inv = 1.0f / s_l[nlo];
  float* og = oLat + ((size_t)bh * 16 + nlo) * RANK_;
  for (int i = 0; i < 4; ++i)
    for (int v = 0; v < 8; ++v)
      og[(wave * 4 + i) * 16 + v + hi * 8] = acc[i][v] * inv;
}

// ---------------------------------------------------------------------------
// K5: outTmp[bt][h*Dh+d] = o_latent[b,h,t,:] @ w_kv_up[:, h*Dh+d]
// ---------------------------------------------------------------------------
__global__ void k_out_up(const float* __restrict__ oLat,
                         const float* __restrict__ wup,
                         float* __restrict__ outTmp) {
  int idx = blockIdx.x * blockDim.x + threadIdx.x;
  if (idx >= 32 * DM_) return;
  int bt = idx / DM_, col = idx % DM_;
  int h = col / Dh_;
  int b = bt / T_, t = bt % T_;
  const float* o = oLat + (((size_t)(b * H_ + h)) * 16 + t) * RANK_;
  const float* w = wup + col;
  float s = 0.f;
  for (int r = 0; r < RANK_; ++r) s += o[r] * w[(size_t)r * DM_];
  outTmp[idx] = s;
}

// ---------------------------------------------------------------------------
// K6: out = outTmp @ w_out
// ---------------------------------------------------------------------------
__global__ void k_final(const float* __restrict__ outTmp,
                        const float* __restrict__ wout,
                        float* __restrict__ out) {
  int idx = blockIdx.x * blockDim.x + threadIdx.x;
  if (idx >= 32 * DM_) return;
  int bt = idx / DM_, m = idx % DM_;
  const float* a = outTmp + (size_t)bt * DM_;
  float s = 0.f;
  for (int k = 0; k < DM_; ++k) s += a[k] * wout[(size_t)k * DM_ + m];
  out[idx] = s;
}

// ---------------------------------------------------------------------------
extern "C" void kernel_launch(void* const* d_in, const int* in_sizes, int n_in,
                              void* d_out, int out_size, void* d_ws, size_t ws_size,
                              hipStream_t stream) {
  (void)in_sizes; (void)n_in; (void)out_size; (void)ws_size;
  const float* x          = (const float*)d_in[0];
  const float* cache_ckv  = (const float*)d_in[1];
  const float* cache_kr   = (const float*)d_in[2];
  const float* w_q        = (const float*)d_in[3];
  const float* w_q_rope   = (const float*)d_in[4];
  const float* w_kv_down  = (const float*)d_in[5];
  const float* w_kv_up    = (const float*)d_in[6];
  const float* w_k_rope   = (const float*)d_in[7];
  const float* w_out      = (const float*)d_in[8];

  char* ws = (char*)d_ws;
  size_t off = 0;
  auto alloc = [&](size_t bytes) { size_t o = off; off = (off + bytes + 255) & ~(size_t)255; return o; };
  float*          qc      = (float*)(ws + alloc((size_t)32 * 1024 * 4));
  float*          qr_raw  = (float*)(ws + alloc((size_t)32 * 1024 * 4));
  float*          ckv_new = (float*)(ws + alloc((size_t)32 * 512 * 4));
  float*          kr_raw  = (float*)(ws + alloc((size_t)32 * 1024 * 4));
  unsigned short* qB      = (unsigned short*)(ws + alloc((size_t)B_ * H_ * 16 * QP_ * 2));
  unsigned short* cKVb    = (unsigned short*)(ws + alloc((size_t)B_ * SPAD_ * RANK_ * 2));
  unsigned short* kRb     = (unsigned short*)(ws + alloc((size_t)B_ * H_ * SPAD_ * R_ * 2));
  float*          oLat    = (float*)(ws + alloc((size_t)B_ * H_ * 16 * RANK_ * 4));
  float*          outTmp  = (float*)(ws + alloc((size_t)32 * DM_ * 4));

  // projections of the 32 new-token rows
  k_proj<<<CDIV(32 * 1024, 256), 256, 0, stream>>>(x, w_q,       qc,      1024);
  k_proj<<<CDIV(32 * 1024, 256), 256, 0, stream>>>(x, w_q_rope,  qr_raw,  1024);
  k_proj<<<CDIV(32 * 512, 256),  256, 0, stream>>>(x, w_kv_down, ckv_new, 512);
  k_proj<<<CDIV(32 * 1024, 256), 256, 0, stream>>>(x, w_k_rope,  kr_raw,  1024);

  // absorbed q (scaled, rope'd, bf16) + bf16 caches
  k_build_q<<<CDIV(B_ * H_ * 16 * QP_, 256), 256, 0, stream>>>(qc, qr_raw, w_kv_up, qB);
  k_build_ckv<<<CDIV(B_ * SPAD_ * (RANK_ / 8), 256), 256, 0, stream>>>(cache_ckv, ckv_new, cKVb);
  k_build_kr<<<CDIV(B_ * H_ * SPAD_ * (R_ / 8), 256), 256, 0, stream>>>(cache_kr, kr_raw, kRb);

  // latent-space flash attention (WMMA bf16 + TDM staging)
  k_mla_attn<<<B_ * H_, 256, LDSB, stream>>>(qB, cKVb, kRb, oLat);

  // output projections
  k_out_up<<<CDIV(32 * DM_, 256), 256, 0, stream>>>(oLat, w_kv_up, outTmp);
  k_final<<<CDIV(32 * DM_, 256), 256, 0, stream>>>(outTmp, w_out, (float*)d_out);
}
